// DecoderLayer_15891378995467
// MI455X (gfx1250) — matrix operand order repaired and checked by reading, not compiled, not run on hardware
//
#include <hip/hip_runtime.h>
#include <cstdint>
#include <cstddef>

typedef __attribute__((ext_vector_type(16))) __bf16 v16bf;
typedef __attribute__((ext_vector_type(8)))  float  v8f;

#define H_  8
#define DH_ 128
#define DM_ 1024
#define SLEN_ 256
#define BSZ_ 64
#define ROWS_ (SLEN_ * BSZ_)          // 16384
#define SCALE_ 0.08838834764831845f   // 1/sqrt(128)
#define LN_EPS_ 1e-5f

// ---- WMMA operand loaders (bf16 16x16x32) -------------------------------
// A operand (16x32, row-major source): lane r(0..15) = row; kh=(lane>>4)*16:
//   16 contiguous K values starting at p.
__device__ __forceinline__ v16bf load_a16(const __bf16* p) {
  v16bf r;
#pragma unroll
  for (int j = 0; j < 16; ++j) r[j] = p[j];
  return r;
}
// B operand (32x16) from [N,K] row-major source: lane n(0..15) = column,
// kh=(lane>>4)*16: 16 contiguous K values starting at p.
__device__ __forceinline__ v16bf load_b16(const __bf16* p) {
  v16bf r;
#pragma unroll
  for (int j = 0; j < 16; ++j) r[j] = p[j];
  return r;
}
__device__ __forceinline__ v8f wmma_bf16(v16bf a, v16bf b, v8f c) {
  return __builtin_amdgcn_wmma_f32_16x16x32_bf16(false, a, false, b, (short)0, c,
                                                 false, false);
}

// ---- GEMM: C[M,N] = A[M,K](f32) @ Bw[N,K](f32)^T, bf16 WMMA, f32 acc ----
// Block tile 128x128x64, 256 threads (8 waves, 4m x 2n), wave tile 32x64.
template <int OUT_BF16>
__global__ __launch_bounds__(256) void gemm_bt_kernel(
    const float* __restrict__ A, const float* __restrict__ Bw,
    void* __restrict__ Cout, int M, int N, int K) {
  __shared__ __bf16 sA[128 * 64];
  __shared__ __bf16 sB[128 * 64];
  const int tid = threadIdx.x;
  const int lane = tid & 31;
  const int wave = tid >> 5;
  const int wm = (wave >> 1) * 32;
  const int wn = (wave & 1) * 64;
  const int m0 = blockIdx.y * 128;
  const int n0 = blockIdx.x * 128;
  const int lr = lane & 15;
  const int kh8 = (lane >> 4) << 3;
  const int kh16 = (lane >> 4) << 4;
  const int r = tid >> 1;
  const int c0 = (tid & 1) * 32;

  v8f acc[2][4];
#pragma unroll
  for (int mi = 0; mi < 2; ++mi)
#pragma unroll
    for (int ni = 0; ni < 4; ++ni) acc[mi][ni] = (v8f)0.0f;

  for (int k0 = 0; k0 < K; k0 += 64) {
    __syncthreads();
    const float* ga = A + (size_t)(m0 + r) * K + k0 + c0;
    const float* gb = Bw + (size_t)(n0 + r) * K + k0 + c0;
#pragma unroll
    for (int j = 0; j < 32; ++j) {
      sA[r * 64 + c0 + j] = (__bf16)ga[j];
      sB[r * 64 + c0 + j] = (__bf16)gb[j];
    }
    __syncthreads();
#pragma unroll
    for (int kk = 0; kk < 2; ++kk) {
      v16bf a[2], bb[4];
#pragma unroll
      for (int mi = 0; mi < 2; ++mi)
        a[mi] = load_a16(&sA[(wm + mi * 16 + lr) * 64 + kk * 32 + kh16]);
#pragma unroll
      for (int ni = 0; ni < 4; ++ni)
        bb[ni] = load_b16(&sB[(wn + ni * 16 + lr) * 64 + kk * 32 + kh16]);
#pragma unroll
      for (int mi = 0; mi < 2; ++mi)
#pragma unroll
        for (int ni = 0; ni < 4; ++ni)
          acc[mi][ni] = wmma_bf16(a[mi], bb[ni], acc[mi][ni]);
    }
  }

  const int rhalf = (lane >> 4) << 3;
#pragma unroll
  for (int mi = 0; mi < 2; ++mi)
#pragma unroll
    for (int ni = 0; ni < 4; ++ni)
#pragma unroll
      for (int i = 0; i < 8; ++i) {
        const int row = m0 + wm + mi * 16 + i + rhalf;
        const int col = n0 + wn + ni * 16 + lr;
        const float v = acc[mi][ni][i];
        if (OUT_BF16)
          ((__bf16*)Cout)[(size_t)row * N + col] = (__bf16)v;
        else
          ((float*)Cout)[(size_t)row * N + col] = v;
      }
}

// ---- phi + repack: qkv[row,4096](bf16) -> q,k1,k2 (phi'd), v in [bh][l][d]
__global__ __launch_bounds__(256) void phi_pack_kernel(
    const __bf16* __restrict__ qkvb, __bf16* __restrict__ qb,
    __bf16* __restrict__ k1b, __bf16* __restrict__ k2b,
    __bf16* __restrict__ vb) {
  const int row = blockIdx.x;          // row = l*BSZ + b
  const int l = row >> 6;
  const int b = row & 63;
  const int tid = threadIdx.x;
  const int chunk = tid >> 3;          // 32 chunks of 128 per row
  const int sub = tid & 7;             // 8 threads per chunk, 16 elems each
  const int head = chunk >> 2;
  const int part = chunk & 3;          // 0:q 1:k1 2:k2 3:v
  const __bf16* src = qkvb + (size_t)row * 4096 + chunk * 128 + sub * 16;

  float y[16];
  float ps = 0.0f;
#pragma unroll
  for (int j = 0; j < 16; ++j) {
    const float x = (float)src[j];
    if (part < 3) {                    // elu(x)+1
      const float e = (x > 0.0f) ? (x + 1.0f) : __expf(x);
      y[j] = e;
      ps += e;
    } else {
      y[j] = x;
    }
  }
  __shared__ float red[32][8];
  __shared__ float inv[32];
  red[chunk][sub] = ps;
  __syncthreads();
  if (sub == 0) {
    float s = 0.0f;
#pragma unroll
    for (int k = 0; k < 8; ++k) s += red[chunk][k];
    inv[chunk] = 1.0f / s;
  }
  __syncthreads();
  const float sc = (part < 3) ? inv[chunk] : 1.0f;
  __bf16* dst = (part == 0) ? qb : (part == 1) ? k1b : (part == 2) ? k2b : vb;
  const size_t o = ((size_t)(b * H_ + head) * SLEN_ + l) * DH_ + sub * 16;
#pragma unroll
  for (int j = 0; j < 16; ++j) dst[o + j] = (__bf16)(y[j] * sc);
}

// ---- causal linear attention, one (b,h) per block, 512 thr = 16 waves ----
// Waves: mc = wave>>1 (32-row stripe of l), nc = wave&1 (64-col stripe of d).
// Register-pressure-aware: only ONE 16x16 S1/S2 tile pair live at a time.
__global__ __launch_bounds__(512, 1) void attn_kernel(
    const __bf16* __restrict__ qb, const __bf16* __restrict__ k1b,
    const __bf16* __restrict__ k2b, const __bf16* __restrict__ vb,
    const float* __restrict__ pi0, float* __restrict__ lof) {
  __shared__ __bf16 sK1[SLEN_ * DH_];     // [s][d]   64KB
  __shared__ __bf16 sK2[SLEN_ * DH_];     // [s][d]   64KB
  __shared__ __bf16 sVT[DH_ * SLEN_];     // [d][s]   64KB
  __shared__ __bf16 sScr[16 * 32 * 33];   // per-wave 32x33 score scratch

  const int bh = blockIdx.x;
  const int b = bh >> 3;
  const int h = bh & 7;
  const size_t base = (size_t)bh * SLEN_ * DH_;
  const int tid = threadIdx.x;
  const int lane = tid & 31;
  const int wave = tid >> 5;
  const int mc = wave >> 1;               // 0..7 -> rows [mc*32, mc*32+32)
  const int nc = wave & 1;                // 0..1 -> cols [nc*64, nc*64+64)
  const int lr = lane & 15;
  const int kh8 = (lane >> 4) << 3;
  const int kh16 = (lane >> 4) << 4;
  const int rhalf = (lane >> 4) << 3;

  // Stage k1,k2 (copy) and v (transpose) into LDS, 16B granules.
  for (int c = tid; c < 4096; c += 512) {
    const int s = c >> 4;
    const int d0 = (c & 15) * 8;
    const __bf16* g1 = k1b + base + s * DH_ + d0;
    const __bf16* g2 = k2b + base + s * DH_ + d0;
    const __bf16* gv = vb + base + s * DH_ + d0;
#pragma unroll
    for (int j = 0; j < 8; ++j) {
      sK1[s * DH_ + d0 + j] = g1[j];
      sK2[s * DH_ + d0 + j] = g2[j];
      sVT[(d0 + j) * SLEN_ + s] = gv[j];
    }
  }
  __syncthreads();

  // q rows for this wave, as WMMA A operands (K=128 in 4 chunks of 32)
  v16bf qa[2][4];
#pragma unroll
  for (int mi = 0; mi < 2; ++mi)
#pragma unroll
    for (int kk = 0; kk < 4; ++kk)
      qa[mi][kk] =
          load_a16(qb + base + (size_t)(mc * 32 + mi * 16 + lr) * DH_ +
                   kk * 32 + kh16);

  // per-row clipped pi (constant across d)
  float pv[2][8];
#pragma unroll
  for (int mi = 0; mi < 2; ++mi)
#pragma unroll
    for (int i = 0; i < 8; ++i) {
      const int l = mc * 32 + mi * 16 + i + rhalf;
      const float p = pi0[h * 256 + l];
      pv[mi][i] = fminf(fmaxf(p, 0.0f), 1.0f);
    }

  v8f o[2][4];
#pragma unroll
  for (int mi = 0; mi < 2; ++mi)
#pragma unroll
    for (int ni = 0; ni < 4; ++ni) o[mi][ni] = (v8f)0.0f;

  __bf16* scr = sScr + wave * (32 * 33);

  for (int scb = 0; scb <= mc; ++scb) {   // causal: only lower-tri 32-chunks
    // One (mi,si) score tile pair at a time: S live set = 16 VGPRs.
#pragma unroll
    for (int mi = 0; mi < 2; ++mi) {
#pragma unroll
      for (int si = 0; si < 2; ++si) {
        v8f s1t = (v8f)0.0f;
        v8f s2t = (v8f)0.0f;
#pragma unroll
        for (int kk = 0; kk < 4; ++kk) {
          const int srow = (scb * 32 + si * 16 + lr) * DH_ + kk * 32 + kh16;
          const v16bf b1 = load_b16(&sK1[srow]);
          s1t = wmma_bf16(qa[mi][kk], b1, s1t);
          const v16bf b2 = load_b16(&sK2[srow]);
          s2t = wmma_bf16(qa[mi][kk], b2, s2t);
        }
        // mask (diagonal chunk) + pi-combine -> bf16 scratch (A-layout feed)
#pragma unroll
        for (int i = 0; i < 8; ++i) {
          const int ll = mi * 16 + i + rhalf;    // local l in [0,32)
          const int ssl = si * 16 + lr;          // local s in [0,32)
          float x1 = s1t[i];
          float x2 = s2t[i];
          if (scb == mc && ssl > ll) { x1 = 0.0f; x2 = 0.0f; }
          const float p = pv[mi][i];
          scr[ll * 33 + ssl] = (__bf16)(p * x1 + (1.0f - p) * x2);
        }
      }
    }
    // O += Sc @ v  (reduction over s-chunk of 32)
#pragma unroll
    for (int mi = 0; mi < 2; ++mi) {
      const v16bf sa = load_a16(&scr[(mi * 16 + lr) * 33 + kh16]);
#pragma unroll
      for (int ni = 0; ni < 4; ++ni) {
        const v16bf vop =
            load_b16(&sVT[(nc * 64 + ni * 16 + lr) * SLEN_ + scb * 32 + kh16]);
        o[mi][ni] = wmma_bf16(sa, vop, o[mi][ni]);
      }
    }
  }

  // layer_out[l*BSZ+b][h*128+d] = SCALE * O
#pragma unroll
  for (int mi = 0; mi < 2; ++mi)
#pragma unroll
    for (int ni = 0; ni < 4; ++ni)
#pragma unroll
      for (int i = 0; i < 8; ++i) {
        const int l = mc * 32 + mi * 16 + i + rhalf;
        const int d = nc * 64 + ni * 16 + lr;
        lof[((size_t)l * BSZ_ + b) * DM_ + h * DH_ + d] = SCALE_ * o[mi][ni][i];
      }
}

// ---- residual + LayerNorm over DM=1024, one row per block ---------------
__global__ __launch_bounds__(256) void ln_kernel(
    const float* __restrict__ hin, const float* __restrict__ attnf,
    const float* __restrict__ gamma, const float* __restrict__ beta,
    float* __restrict__ out) {
  const int row = blockIdx.x;
  const int tid = threadIdx.x;
  __shared__ float rs[256];
  __shared__ float rq[256];
  float x[4];
  float s = 0.0f, q = 0.0f;
#pragma unroll
  for (int k = 0; k < 4; ++k) {
    const int c = tid + k * 256;
    const float v = hin[(size_t)row * DM_ + c] + attnf[(size_t)row * DM_ + c];
    x[k] = v;
    s += v;
    q += v * v;
  }
  rs[tid] = s;
  rq[tid] = q;
  __syncthreads();
  for (int off = 128; off > 0; off >>= 1) {
    if (tid < off) {
      rs[tid] += rs[tid + off];
      rq[tid] += rq[tid + off];
    }
    __syncthreads();
  }
  const float mu = rs[0] * (1.0f / DM_);
  const float var = rq[0] * (1.0f / DM_) - mu * mu;
  const float inv = rsqrtf(var + LN_EPS_);
#pragma unroll
  for (int k = 0; k < 4; ++k) {
    const int c = tid + k * 256;
    out[(size_t)row * DM_ + c] = gamma[c] * (x[k] - mu) * inv + beta[c];
  }
}

extern "C" void kernel_launch(void* const* d_in, const int* in_sizes, int n_in,
                              void* d_out, int out_size, void* d_ws,
                              size_t ws_size, hipStream_t stream) {
  (void)in_sizes; (void)n_in; (void)out_size; (void)ws_size;
  const float* h     = (const float*)d_in[0];
  const float* W_qkv = (const float*)d_in[1];
  const float* W_o   = (const float*)d_in[2];
  const float* pi0   = (const float*)d_in[3];
  const float* g     = (const float*)d_in[4];
  const float* beta  = (const float*)d_in[5];

  uint8_t* w = (uint8_t*)d_ws;
  __bf16* qkvb = (__bf16*)(w);                         // 16384*4096*2 = 128MB
  __bf16* qb   = (__bf16*)(w + 134217728ull);          // 512*256*128*2 each
  __bf16* k1b  = (__bf16*)(w + 167772160ull);
  __bf16* k2b  = (__bf16*)(w + 201326592ull);
  __bf16* vb   = (__bf16*)(w + 234881024ull);
  float*  lof  = (float*)(w + 268435456ull);           // 16384*1024*4 = 64MB
  float*  attnf= (float*)(w + 335544320ull);           // 64MB

  // 1) qkv = bf16( h @ W_qkv^T )
  gemm_bt_kernel<1><<<dim3(4096 / 128, ROWS_ / 128), 256, 0, stream>>>(
      h, W_qkv, (void*)qkvb, ROWS_, 4096, DM_);
  // 2) phi + repack into per-head [bh][l][d]
  phi_pack_kernel<<<ROWS_, 256, 0, stream>>>(qkvb, qb, k1b, k2b, vb);
  // 3) dual causal linear attention, pi-combined on scores
  attn_kernel<<<H_ * BSZ_, 512, 0, stream>>>(qb, k1b, k2b, vb, pi0, lof);
  // 4) attn_out = layer_out @ W_o^T
  gemm_bt_kernel<0><<<dim3(DM_ / 128, ROWS_ / 128), 256, 0, stream>>>(
      lof, W_o, (void*)attnf, ROWS_, DM_, DM_);
  // 5) residual + LayerNorm
  ln_kernel<<<ROWS_, 256, 0, stream>>>(h, attnf, g, beta, (float*)d_out);
}
